// Method_GCN_Cora_32882269618962
// MI455X (gfx1250) — matrix operand order repaired and checked by reading, not compiled, not run on hardware
//
#include <hip/hip_runtime.h>
#include <hip/hip_bf16.h>

typedef __attribute__((ext_vector_type(16))) __bf16 v16bf;
typedef __attribute__((ext_vector_type(8)))  __bf16 v8bf;
typedef __attribute__((ext_vector_type(8)))  float  v8f;

#define NNODES 100000
#define FIN    1433
#define H1DIM  200
#define H2DIM  80
#define CDIM   7

#define ROWT   (NNODES/16)   // 6250 row tiles
#define H1T    13            // col tiles for 200 -> 208
#define K1C    45            // K chunks: 45*32 = 1440 >= 1433
#define Y1S    208           // Y1 padded stride
#define H1BS   224           // H1 bf16 padded K stride (7*32)
#define H2T    5             // col tiles: 5*16 = 80
#define K2C    7             // K chunks: 7*32 = 224 >= 200

__device__ inline unsigned short f2bf_bits(float f) {
    __bf16 h = (__bf16)f;
    return __builtin_bit_cast(unsigned short, h);
}

// ---- pack W1 [1433 x 200] f32 -> bf16 B-operand layout [13][45][32 lanes][8 dwords]
__global__ void pack_w1_kernel(const float* __restrict__ W1, unsigned* __restrict__ Bp) {
    int idx = blockIdx.x * blockDim.x + threadIdx.x;
    const int total = H1T * K1C * 32 * 8;
    if (idx >= total) return;
    int d    = idx & 7;
    int lane = (idx >> 3) & 31;
    int rest = idx >> 8;            // ct*K1C + kt
    int kt   = rest % K1C;
    int ct   = rest / K1C;
    int colg = ct * 16 + (lane & 15);
    int k0   = kt * 32 + ((lane >> 4) * 16) + 2 * d;
    unsigned lo = 0, hi = 0;
    if (colg < H1DIM) {
        if (k0     < FIN) lo = f2bf_bits(W1[(size_t)k0 * H1DIM + colg]);
        if (k0 + 1 < FIN) hi = f2bf_bits(W1[(size_t)(k0 + 1) * H1DIM + colg]);
    }
    Bp[idx] = lo | (hi << 16);
}

// ---- pack W2 [200 x 80] f32 -> bf16 B-operand layout [5][7][32][8]
__global__ void pack_w2_kernel(const float* __restrict__ W2, unsigned* __restrict__ Bp) {
    int idx = blockIdx.x * blockDim.x + threadIdx.x;
    const int total = H2T * K2C * 32 * 8;
    if (idx >= total) return;
    int d    = idx & 7;
    int lane = (idx >> 3) & 31;
    int rest = idx >> 8;            // ct*K2C + kt
    int kt   = rest % K2C;
    int ct   = rest / K2C;
    int colg = ct * 16 + (lane & 15);
    int k0   = kt * 32 + ((lane >> 4) * 16) + 2 * d;
    unsigned lo = 0, hi = 0;
    if (colg < H2DIM) {
        if (k0     < H1DIM) lo = f2bf_bits(W2[(size_t)k0 * H2DIM + colg]);
        if (k0 + 1 < H1DIM) hi = f2bf_bits(W2[(size_t)(k0 + 1) * H2DIM + colg]);
    }
    Bp[idx] = lo | (hi << 16);
}

// ---- GEMM1: Y1[100000 x 208] = x[100000 x 1433] @ W1 (bf16 WMMA, f32 acc)
// One wave per 16-row tile; computes all 13 column tiles so x is read once.
__global__ __launch_bounds__(256)
void gemm1_kernel(const float* __restrict__ x,
                  const unsigned* __restrict__ Bp,
                  float* __restrict__ Y1) {
    const int lane = threadIdx.x & 31;
    const int rt   = blockIdx.x * (blockDim.x >> 5) + (threadIdx.x >> 5);
    if (rt >= ROWT) return;                      // wave-uniform
    const int col = lane & 15;
    const int kb  = (lane >> 4) * 16;            // A: K sub-offset for this half-wave
    const float* xrow = x + (size_t)(rt * 16 + col) * FIN;

    v8f acc[H1T];
    #pragma unroll
    for (int t = 0; t < H1T; ++t)
        #pragma unroll
        for (int j = 0; j < 8; ++j) acc[t][j] = 0.0f;

    for (int kt = 0; kt < K1C; ++kt) {
        const int k0 = kt * 32;
        __builtin_prefetch(xrow + k0 + 128, 0, 0);   // global_prefetch_b8
        v16bf a;
        if (k0 + 32 <= FIN) {
            #pragma unroll
            for (int i = 0; i < 8; ++i) {
                a[i]     = (__bf16)xrow[k0 + kb + i];
                a[8 + i] = (__bf16)xrow[k0 + 8 + kb + i];
            }
        } else {                                     // last chunk, K guard
            #pragma unroll
            for (int i = 0; i < 8; ++i) {
                int ka = k0 + kb + i;
                int kc = k0 + 8 + kb + i;
                a[i]     = (ka < FIN) ? (__bf16)xrow[ka] : (__bf16)0.0f;
                a[8 + i] = (kc < FIN) ? (__bf16)xrow[kc] : (__bf16)0.0f;
            }
        }
        #pragma unroll
        for (int ct = 0; ct < H1T; ++ct) {
            v16bf b = *reinterpret_cast<const v16bf*>(
                Bp + (((size_t)ct * K1C + kt) * 32 + lane) * 8);
            acc[ct] = __builtin_amdgcn_wmma_f32_16x16x32_bf16(
                false, a, false, b, (short)0, acc[ct], false, false);
        }
    }
    const int mhi = (lane >> 4) * 8;
    #pragma unroll
    for (int ct = 0; ct < H1T; ++ct) {
        float* yb = Y1 + (size_t)(rt * 16) * Y1S + ct * 16 + col;
        #pragma unroll
        for (int j = 0; j < 8; ++j)
            yb[(size_t)(j + mhi) * Y1S] = acc[ct][j];
    }
}

// ---- GEMM2: Y2[100000 x 80] = H1b[100000 x 224 bf16] @ W2
__global__ __launch_bounds__(256)
void gemm2_kernel(const __bf16* __restrict__ Hb,
                  const unsigned* __restrict__ Bp,
                  float* __restrict__ Y2) {
    const int lane = threadIdx.x & 31;
    const int rt   = blockIdx.x * (blockDim.x >> 5) + (threadIdx.x >> 5);
    if (rt >= ROWT) return;
    const int col = lane & 15;
    const int kb  = (lane >> 4) * 16;
    const __bf16* hrow = Hb + (size_t)(rt * 16 + col) * H1BS;

    v8f acc[H2T];
    #pragma unroll
    for (int t = 0; t < H2T; ++t)
        #pragma unroll
        for (int j = 0; j < 8; ++j) acc[t][j] = 0.0f;

    #pragma unroll
    for (int kt = 0; kt < K2C; ++kt) {
        const int k0 = kt * 32;
        v8bf lo = *reinterpret_cast<const v8bf*>(hrow + k0 + kb);        // 16B aligned
        v8bf hi = *reinterpret_cast<const v8bf*>(hrow + k0 + 8 + kb);
        v16bf a;
        #pragma unroll
        for (int i = 0; i < 8; ++i) { a[i] = lo[i]; a[8 + i] = hi[i]; }
        #pragma unroll
        for (int ct = 0; ct < H2T; ++ct) {
            v16bf b = *reinterpret_cast<const v16bf*>(
                Bp + (((size_t)ct * K2C + kt) * 32 + lane) * 8);
            acc[ct] = __builtin_amdgcn_wmma_f32_16x16x32_bf16(
                false, a, false, b, (short)0, acc[ct], false, false);
        }
    }
    const int mhi = (lane >> 4) * 8;
    #pragma unroll
    for (int ct = 0; ct < H2T; ++ct) {
        float* yb = Y2 + (size_t)(rt * 16) * H2DIM + ct * 16 + col;
        #pragma unroll
        for (int j = 0; j < 8; ++j)
            yb[(size_t)(j + mhi) * H2DIM] = acc[ct][j];
    }
}

// ---- zero fill
__global__ void zero_f32_kernel(float* __restrict__ p, long long n) {
    long long i = (long long)blockIdx.x * blockDim.x + threadIdx.x;
    if (i < n) p[i] = 0.0f;
}

// ---- SpMM: one wave per edge; Hacc[dst] += w * Yin[src] (L2-resident gathers)
__global__ void spmm_kernel(const float* __restrict__ Yin, int stride_in,
                            const int* __restrict__ esrc, const int* __restrict__ edst,
                            const float* __restrict__ ew,
                            float* __restrict__ Hacc, int ncols, int E) {
    int gtid = blockIdx.x * blockDim.x + threadIdx.x;
    int e    = gtid >> 5;
    int lane = gtid & 31;
    if (e >= E) return;
    int   s = esrc[e];
    int   d = edst[e];
    float w = ew[e];
    const float* yr = Yin + (size_t)s * stride_in;
    float*       hr = Hacc + (size_t)d * ncols;
    for (int c = lane; c < ncols; c += 32)
        atomicAdd(&hr[c], w * yr[c]);
}

// ---- bias + ReLU, write bf16 padded copy for GEMM2
__global__ void bias_relu1_kernel(const float* __restrict__ Hacc,
                                  const float* __restrict__ b1,
                                  __bf16* __restrict__ Hb) {
    int i = blockIdx.x * blockDim.x + threadIdx.x;
    if (i >= NNODES * H1BS) return;
    int n = i / H1BS, c = i % H1BS;
    float v = 0.0f;
    if (c < H1DIM) {
        v = Hacc[(size_t)n * H1DIM + c] + b1[c];
        v = v > 0.0f ? v : 0.0f;
    }
    Hb[i] = (__bf16)v;
}

// ---- bias + ReLU, f32 out
__global__ void bias_relu2_kernel(const float* __restrict__ Hacc,
                                  const float* __restrict__ b2,
                                  float* __restrict__ H2) {
    int i = blockIdx.x * blockDim.x + threadIdx.x;
    if (i >= NNODES * H2DIM) return;
    int c = i % H2DIM;
    float v = Hacc[i] + b2[c];
    H2[i] = v > 0.0f ? v : 0.0f;
}

// ---- fc: [100000 x 80] @ [80 x 7] + bias, one thread per node
__global__ void fc_kernel(const float* __restrict__ H2,
                          const float* __restrict__ Wfc,
                          const float* __restrict__ bfc,
                          float* __restrict__ out) {
    int n = blockIdx.x * blockDim.x + threadIdx.x;
    if (n >= NNODES) return;
    float acc[CDIM];
    #pragma unroll
    for (int c = 0; c < CDIM; ++c) acc[c] = bfc[c];
    const float* hr = H2 + (size_t)n * H2DIM;
    for (int k = 0; k < H2DIM; ++k) {
        float h = hr[k];
        #pragma unroll
        for (int c = 0; c < CDIM; ++c) acc[c] += h * Wfc[k * CDIM + c];
    }
    #pragma unroll
    for (int c = 0; c < CDIM; ++c) out[(size_t)n * CDIM + c] = acc[c];
}

extern "C" void kernel_launch(void* const* d_in, const int* in_sizes, int n_in,
                              void* d_out, int out_size, void* d_ws, size_t ws_size,
                              hipStream_t stream) {
    const float* x    = (const float*)d_in[0];
    const int*   esrc = (const int*)d_in[1];
    const int*   edst = (const int*)d_in[2];
    const float* ew   = (const float*)d_in[3];
    const float* W1   = (const float*)d_in[4];
    const float* b1   = (const float*)d_in[5];
    const float* W2   = (const float*)d_in[6];
    const float* b2   = (const float*)d_in[7];
    const float* Wfc  = (const float*)d_in[8];
    const float* bfc  = (const float*)d_in[9];
    float* out = (float*)d_out;
    const int E = in_sizes[1];

    char* ws = (char*)d_ws;
    size_t off = 0;
    auto take = [&](size_t bytes) -> char* {
        char* p = ws + off;
        off += (bytes + 255) & ~(size_t)255;
        return p;
    };
    unsigned* B1p     = (unsigned*)take((size_t)H1T * K1C * 32 * 8 * 4);
    unsigned* B2p     = (unsigned*)take((size_t)H2T * K2C * 32 * 8 * 4);
    float*    regionA = (float*)take((size_t)NNODES * Y1S * 4);   // Y1, then Y2
    float*    regionB = (float*)take((size_t)NNODES * H1DIM * 4); // H1acc, then H2acc+H2
    __bf16*   H1b     = (__bf16*)take((size_t)NNODES * H1BS * 2);

    float* Y1    = regionA;
    float* H1acc = regionB;
    float* Y2    = regionA;                        // Y1 dead after spmm1
    float* H2acc = regionB;                        // H1acc dead after bias_relu1
    float* H2    = regionB + (size_t)NNODES * H2DIM;

    // Layer 1
    pack_w1_kernel<<<(H1T * K1C * 256 + 255) / 256, 256, 0, stream>>>(W1, B1p);
    gemm1_kernel<<<(ROWT + 7) / 8, 256, 0, stream>>>(x, B1p, Y1);
    zero_f32_kernel<<<((long long)NNODES * H1DIM + 255) / 256, 256, 0, stream>>>(
        H1acc, (long long)NNODES * H1DIM);
    spmm_kernel<<<(E * 32 + 255) / 256, 256, 0, stream>>>(
        Y1, Y1S, esrc, edst, ew, H1acc, H1DIM, E);
    bias_relu1_kernel<<<(NNODES * H1BS + 255) / 256, 256, 0, stream>>>(H1acc, b1, H1b);

    // Layer 2
    pack_w2_kernel<<<(H2T * K2C * 256 + 255) / 256, 256, 0, stream>>>(W2, B2p);
    gemm2_kernel<<<(ROWT + 7) / 8, 256, 0, stream>>>(H1b, B2p, Y2);
    zero_f32_kernel<<<((long long)NNODES * H2DIM + 255) / 256, 256, 0, stream>>>(
        H2acc, (long long)NNODES * H2DIM);
    spmm_kernel<<<(E * 32 + 255) / 256, 256, 0, stream>>>(
        Y2, H2DIM, esrc, edst, ew, H2acc, H2DIM, E);
    bias_relu2_kernel<<<(NNODES * H2DIM + 255) / 256, 256, 0, stream>>>(H2acc, b2, H2);

    // FC head
    fc_kernel<<<(NNODES + 255) / 256, 256, 0, stream>>>(H2, Wfc, bfc, out);
}
